// LocalCrossAttention_644245095138
// MI455X (gfx1250) — matrix operand order repaired and checked by reading, not compiled, not run on hardware
//
#include <hip/hip_runtime.h>

#define D_DIM 1024
#define N_DIM 8192

// LDS layout for attention kernel (dynamic shared memory)
#define QL_STRIDE 1032               // 64 rows x 1024 bf16, padded (+8) -> conflict-free ds_load_b128
#define QL_ELEMS  (64 * QL_STRIDE)
#define QL_BYTES  (QL_ELEMS * 2)     // 132096
#define PL_STRIDE 136                // 64 x 128 bf16 probs tile, padded
#define PL_BYTES  (64 * PL_STRIDE * 2)  // 17408
#define ATTN_LDS_BYTES (QL_BYTES + PL_BYTES)

typedef __bf16 bf16;
typedef __attribute__((ext_vector_type(16))) __bf16 v16bf;
typedef __attribute__((ext_vector_type(8)))  __bf16 v8bf;
typedef __attribute__((ext_vector_type(4)))  __bf16 v4bf;
typedef __attribute__((ext_vector_type(8)))  float  v8f;

// ---------------------------------------------------------------------------
// WMMA helper (gfx1250: wave32, V_WMMA_F32_16X16X32_BF16)
// ---------------------------------------------------------------------------
__device__ __forceinline__ v8f wmma_bf16(v16bf a, v16bf b, v8f c) {
  return __builtin_amdgcn_wmma_f32_16x16x32_bf16(false, a, false, b,
                                                 (short)0, c, false, false);
}

// A-matrix fragment, 16x32 bf16, row-major source with leading dim ld.
// lanes 0-15: row M=lane, K chunks {0..7, 16..23}; lanes 16-31: {8..15, 24..31}
__device__ __forceinline__ v16bf load_a_frag(const bf16* __restrict__ src,
                                             int ld, int rbase, int kbase) {
  const int lane = threadIdx.x & 31;
  const int lo = lane & 15, hi = lane >> 4;
  const bf16* row = src + (size_t)(rbase + lo) * ld + kbase + hi * 8;
  v8bf c0 = *(const v8bf*)(row);
  v8bf c1 = *(const v8bf*)(row + 16);
  return __builtin_shufflevector(c0, c1, 0, 1, 2, 3, 4, 5, 6, 7,
                                 8, 9, 10, 11, 12, 13, 14, 15);
}

// B-matrix fragment, 32x16 bf16, where B(k,n) = src[rbase+n][kbase+k]
// (source holds columns of B as contiguous rows — K^T rows or V^T rows)
// lanes 0-15: N=lane, K chunks {0..7, 16..23}; lanes 16-31: N=lane-16, {8..15, 24..31}
__device__ __forceinline__ v16bf load_b_frag(const bf16* __restrict__ src,
                                             int ld, int rbase, int kbase) {
  const int lane = threadIdx.x & 31;
  const int lo = lane & 15, hi = lane >> 4;
  const bf16* row = src + (size_t)(rbase + lo) * ld + kbase + hi * 8;
  v8bf c0 = *(const v8bf*)(row);
  v8bf c1 = *(const v8bf*)(row + 16);
  return __builtin_shufflevector(c0, c1, 0, 1, 2, 3, 4, 5, 6, 7,
                                 8, 9, 10, 11, 12, 13, 14, 15);
}

// ---------------------------------------------------------------------------
// f32 -> bf16 convert (vectorized x4)
// ---------------------------------------------------------------------------
__global__ void cvt_f32_bf16_kernel(const float* __restrict__ in,
                                    bf16* __restrict__ out, int n4) {
  int i = blockIdx.x * blockDim.x + threadIdx.x;
  if (i < n4) {
    float4 f = ((const float4*)in)[i];
    v4bf o;
    o[0] = (bf16)f.x; o[1] = (bf16)f.y; o[2] = (bf16)f.z; o[3] = (bf16)f.w;
    ((v4bf*)out)[i] = o;
  }
}

// ---------------------------------------------------------------------------
// W[k][d] (f32, DxD) -> WT[d][k] (bf16, DxD)
// ---------------------------------------------------------------------------
__global__ void transpose_cvt_kernel(const float* __restrict__ W,
                                     bf16* __restrict__ WT) {
  __shared__ float tile[32][33];
  const int d0 = blockIdx.x * 32, k0 = blockIdx.y * 32;
  tile[threadIdx.y][threadIdx.x] =
      W[(size_t)(k0 + threadIdx.y) * D_DIM + d0 + threadIdx.x];
  __syncthreads();
  WT[(size_t)(d0 + threadIdx.y) * D_DIM + k0 + threadIdx.x] =
      (bf16)tile[threadIdx.x][threadIdx.y];
}

// ---------------------------------------------------------------------------
// Projection: OUT = X @ W + b, via X(bf16) and WT(bf16, pre-transposed).
// Writes row-major bf16 (q/k) and/or transposed bf16 (v).
// grid: (N/64, D/128), block: 256 (8 waves). Wave w owns 16 output columns.
// ---------------------------------------------------------------------------
__global__ __launch_bounds__(256)
void proj_kernel(const bf16* __restrict__ X, const bf16* __restrict__ WT,
                 const float* __restrict__ bias,
                 bf16* __restrict__ outRM, bf16* __restrict__ outTR) {
  const int wave = threadIdx.x >> 5;
  const int lane = threadIdx.x & 31;
  const int lo = lane & 15, hi = lane >> 4;
  const int mb = blockIdx.x * 64;
  const int db = blockIdx.y * 128 + wave * 16;

  v8f acc[4] = {v8f{}, v8f{}, v8f{}, v8f{}};
  for (int kk = 0; kk < D_DIM; kk += 32) {
    v16bf bfrag = load_b_frag(WT, D_DIM, db, kk);
#pragma unroll
    for (int mt = 0; mt < 4; ++mt) {
      v16bf afrag = load_a_frag(X, D_DIM, mb + mt * 16, kk);
      acc[mt] = wmma_bf16(afrag, bfrag, acc[mt]);
    }
  }
  const int dcol = db + lo;
  const float bv = bias[dcol];
#pragma unroll
  for (int mt = 0; mt < 4; ++mt) {
#pragma unroll
    for (int r = 0; r < 8; ++r) {
      const int m = mb + mt * 16 + hi * 8 + r;
      const float v = acc[mt][r] + bv;
      if (outRM) outRM[(size_t)m * D_DIM + dcol] = (bf16)v;
      if (outTR) outTR[(size_t)dcol * N_DIM + m] = (bf16)v;
    }
  }
}

// ---------------------------------------------------------------------------
// Fused sigmoid cross-attention, both directions in one launch:
//   probs = sigmoid(scale * Q @ K^T)   (written f32)
//   ctx   = probs @ V                  (V given transposed: VT[d][n])
// grid: (N/64, 2), block 256 (8 waves). blockIdx.y selects direction.
// Q row-block (64x1024 bf16) is staged once in LDS (padded stride) and
// re-read from LDS every L-tile instead of hammering L2 64x.
// ---------------------------------------------------------------------------
__global__ __launch_bounds__(256)
void attn_kernel(const bf16* __restrict__ Qa, const bf16* __restrict__ Ka,
                 const bf16* __restrict__ VTa, float* __restrict__ probsa,
                 float* __restrict__ ctxa,
                 const bf16* __restrict__ Qb, const bf16* __restrict__ Kb,
                 const bf16* __restrict__ VTb, float* __restrict__ probsb,
                 float* __restrict__ ctxb) {
  extern __shared__ char smem_raw[];
  bf16* Ql = (bf16*)smem_raw;                 // [64][QL_STRIDE]
  bf16* Pl = (bf16*)(smem_raw + QL_BYTES);    // [64][PL_STRIDE]

  const bool dir = (blockIdx.y != 0);
  const bf16* __restrict__ Q     = dir ? Qb : Qa;
  const bf16* __restrict__ K     = dir ? Kb : Ka;
  const bf16* __restrict__ VT    = dir ? VTb : VTa;
  float* __restrict__     probs  = dir ? probsb : probsa;
  float* __restrict__     ctx    = dir ? ctxb : ctxa;

  const int wave = threadIdx.x >> 5;
  const int lane = threadIdx.x & 31;
  const int lo = lane & 15, hi = lane >> 4;
  const int mb = blockIdx.x * 64;
  const float scale = 0.03125f;  // 1/sqrt(1024)

  // ---- stage Q block into LDS (64 rows x 1024 bf16, 16B chunks) ----
  for (int idx = threadIdx.x; idx < 64 * 128; idx += 256) {
    const int r = idx >> 7, c = (idx & 127) * 8;
    *(v8bf*)(Ql + r * QL_STRIDE + c) =
        *(const v8bf*)(Q + (size_t)(mb + r) * D_DIM + c);
  }
  __syncthreads();

  v8f cacc[4][8];
#pragma unroll
  for (int mt = 0; mt < 4; ++mt)
#pragma unroll
    for (int dt = 0; dt < 8; ++dt) cacc[mt][dt] = v8f{};

  for (int Lb = 0; Lb < N_DIM; Lb += 128) {
    // ---- phase 1: S = Q @ K^T for this wave's 16 key-columns ----
    const int nb = Lb + wave * 16;
    v8f s[4] = {v8f{}, v8f{}, v8f{}, v8f{}};
    for (int kk = 0; kk < D_DIM; kk += 32) {
      v16bf bfrag = load_b_frag(K, D_DIM, nb, kk);
#pragma unroll
      for (int mt = 0; mt < 4; ++mt) {
        // A fragment from LDS-staged Q
        const bf16* row = Ql + (mt * 16 + lo) * QL_STRIDE + kk + hi * 8;
        v8bf c0 = *(const v8bf*)(row);
        v8bf c1 = *(const v8bf*)(row + 16);
        v16bf afrag = __builtin_shufflevector(c0, c1, 0, 1, 2, 3, 4, 5, 6, 7,
                                              8, 9, 10, 11, 12, 13, 14, 15);
        s[mt] = wmma_bf16(afrag, bfrag, s[mt]);
      }
    }

    __syncthreads();  // previous iteration's P readers are done
    // ---- sigmoid epilogue: write probs (f32, global) + P (bf16, LDS) ----
    const int col = nb + lo;
#pragma unroll
    for (int mt = 0; mt < 4; ++mt) {
#pragma unroll
      for (int r = 0; r < 8; ++r) {
        const int m = mt * 16 + hi * 8 + r;
        const float p = 1.0f / (1.0f + __expf(-s[mt][r] * scale));
        probs[(size_t)(mb + m) * N_DIM + col] = p;
        Pl[m * PL_STRIDE + wave * 16 + lo] = (bf16)p;
      }
    }
    __syncthreads();

    // ---- phase 2: ctx(64 x 128-slab) += P(64x128) @ V(128 x d-slab) ----
    const int dwb = wave * 128;
#pragma unroll
    for (int l0 = 0; l0 < 128; l0 += 32) {
      v16bf ap[4];
#pragma unroll
      for (int mt = 0; mt < 4; ++mt) {
        const bf16* row = Pl + (mt * 16 + lo) * PL_STRIDE + l0 + hi * 8;
        v8bf c0 = *(const v8bf*)(row);
        v8bf c1 = *(const v8bf*)(row + 16);
        ap[mt] = __builtin_shufflevector(c0, c1, 0, 1, 2, 3, 4, 5, 6, 7,
                                         8, 9, 10, 11, 12, 13, 14, 15);
      }
#pragma unroll
      for (int dt = 0; dt < 8; ++dt) {
        v16bf bv = load_b_frag(VT, N_DIM, dwb + dt * 16, Lb + l0);
#pragma unroll
        for (int mt = 0; mt < 4; ++mt)
          cacc[mt][dt] = wmma_bf16(ap[mt], bv, cacc[mt][dt]);
      }
    }
  }

  // ---- write ctx (f32) ----
#pragma unroll
  for (int mt = 0; mt < 4; ++mt) {
#pragma unroll
    for (int dt = 0; dt < 8; ++dt) {
      const int dcol = wave * 128 + dt * 16 + lo;
#pragma unroll
      for (int r = 0; r < 8; ++r) {
        const int m = mb + mt * 16 + hi * 8 + r;
        ctx[(size_t)m * D_DIM + dcol] = cacc[mt][dt][r];
      }
    }
  }
}

// ---------------------------------------------------------------------------
// Host-side orchestration
// ---------------------------------------------------------------------------
extern "C" void kernel_launch(void* const* d_in, const int* in_sizes, int n_in,
                              void* d_out, int out_size, void* d_ws, size_t ws_size,
                              hipStream_t stream) {
  (void)in_sizes; (void)n_in; (void)out_size; (void)ws_size;
  const float* x1 = (const float*)d_in[0];
  const float* x2 = (const float*)d_in[1];
  // order: Wq1,bq1, Wk1,bk1, Wv1,bv1, Wq2,bq2, Wk2,bk2, Wv2,bv2
  const float* W[6]; const float* B[6];
  for (int i = 0; i < 6; ++i) { W[i] = (const float*)d_in[2 + 2 * i];
                                B[i] = (const float*)d_in[3 + 2 * i]; }

  const size_t ND = (size_t)N_DIM * D_DIM;   // 8 Mi elems
  const size_t DD = (size_t)D_DIM * D_DIM;   // 1 Mi elems
  const size_t NN = (size_t)N_DIM * N_DIM;   // 64 Mi elems

  // workspace layout (bf16 operands; ~140 MB total)
  char* ws = (char*)d_ws;
  size_t off = 0;
  auto take = [&](size_t bytes) -> void* {
    void* p = ws + off; off += (bytes + 255) & ~(size_t)255; return p;
  };
  bf16* x1b  = (bf16*)take(ND * 2);
  bf16* x2b  = (bf16*)take(ND * 2);
  bf16* WT[6];
  for (int i = 0; i < 6; ++i) WT[i] = (bf16*)take(DD * 2);
  bf16* q1b  = (bf16*)take(ND * 2);
  bf16* k1b  = (bf16*)take(ND * 2);
  bf16* v1tb = (bf16*)take(ND * 2);
  bf16* q2b  = (bf16*)take(ND * 2);
  bf16* k2b  = (bf16*)take(ND * 2);
  bf16* v2tb = (bf16*)take(ND * 2);

  // 1) convert inputs to bf16
  {
    const int n4 = (int)(ND / 4);
    const int blocks = (n4 + 255) / 256;
    cvt_f32_bf16_kernel<<<blocks, 256, 0, stream>>>(x1, x1b, n4);
    cvt_f32_bf16_kernel<<<blocks, 256, 0, stream>>>(x2, x2b, n4);
  }
  // 2) transpose+convert weights
  {
    dim3 g(D_DIM / 32, D_DIM / 32), b(32, 32);
    for (int i = 0; i < 6; ++i)
      transpose_cvt_kernel<<<g, b, 0, stream>>>(W[i], WT[i]);
  }
  // 3) projections (q/k row-major; v transposed)
  {
    dim3 g(N_DIM / 64, D_DIM / 128);
    proj_kernel<<<g, 256, 0, stream>>>(x1b, WT[0], B[0], q1b,  nullptr); // q1
    proj_kernel<<<g, 256, 0, stream>>>(x1b, WT[1], B[1], k1b,  nullptr); // k1
    proj_kernel<<<g, 256, 0, stream>>>(x1b, WT[2], B[2], nullptr, v1tb); // v1^T
    proj_kernel<<<g, 256, 0, stream>>>(x2b, WT[3], B[3], q2b,  nullptr); // q2
    proj_kernel<<<g, 256, 0, stream>>>(x2b, WT[4], B[4], k2b,  nullptr); // k2
    proj_kernel<<<g, 256, 0, stream>>>(x2b, WT[5], B[5], nullptr, v2tb); // v2^T
  }
  // 4) fused attention, both directions in one grid
  // d_out layout (floats): ctx2 | probs2 | ctx1 | probs1
  float* out    = (float*)d_out;
  float* ctx2   = out;
  float* probs2 = out + ND;
  float* ctx1   = probs2 + NN;
  float* probs1 = ctx1 + ND;
  {
    static bool attr_set = false;
    // Allow >64KB dynamic LDS (gfx1250 WGP has 320KB; we use ~146KB/WG).
    hipFuncSetAttribute((const void*)attn_kernel,
                        hipFuncAttributeMaxDynamicSharedMemorySize,
                        ATTN_LDS_BYTES);
    (void)attr_set;
    dim3 g(N_DIM / 64, 2);
    attn_kernel<<<g, 256, ATTN_LDS_BYTES, stream>>>(
        q1b, k2b, v2tb, probs2, ctx2,   // dir 0: q1 vs k2/v2 -> probs2/ctx2
        q2b, k1b, v1tb, probs1, ctx1);  // dir 1: q2 vs k1/v1 -> probs1/ctx1
  }
}
